// SS2D_10496900072064
// MI455X (gfx1250) — compile-verified, matrix operand order checked by reading, not run on hardware
//
#include <hip/hip_runtime.h>
#include <hip/hip_bf16.h>

// ---------------- problem constants ----------------
#define DM    256     // D_MODEL
#define DI    512     // D_INNER
#define NSTT  16      // D_STATE
#define DTR   16      // DT_RANK
#define XDW   48      // DT_RANK + 2*D_STATE
#define HH    48
#define WWD   48
#define LSEQ  2304    // 48*48
#define NB    4       // batch
#define NSEQ  16      // 4 dirs * 4 batch
#define MROW  (NSEQ*LSEQ)   // 36864
#define NCH   16      // scan chunks
#define CHUNK 144     // LSEQ / NCH
#define PFD   8       // scan software-prefetch distance (rows)

typedef __bf16          v16bf __attribute__((ext_vector_type(16)));
typedef float           v8f   __attribute__((ext_vector_type(8)));
typedef unsigned short  v8u16 __attribute__((ext_vector_type(8)));

union BFfrag { v16bf bf; v8u16 h[2]; };

__device__ __forceinline__ unsigned short f2bf(float f) {
  unsigned u = __float_as_uint(f);
  unsigned r = u + 0x7FFFu + ((u >> 16) & 1u);   // round-to-nearest-even
  return (unsigned short)(r >> 16);
}
__device__ __forceinline__ float siluf(float v) {
  return v * (1.0f / (1.0f + __expf(-v)));
}

// ---------------- 0) gather 4 directional sequences, convert to bf16 ----------------
__global__ __launch_bounds__(256) void gather_xs(const float* __restrict__ x,
                                                 unsigned short* __restrict__ xs_bf) {
  int idx = blockIdx.x * 256 + threadIdx.x;            // NSEQ*LSEQ*DM threads
  int c   = idx & (DM - 1);
  int t   = idx >> 8;
  int l   = t % LSEQ;
  int seq = t / LSEQ;
  int b   = seq & 3;
  int dir = seq >> 2;
  int i = l / WWD, j = l % WWD;
  int hh, ww;
  if      (dir == 0) { hh = i;      ww = j;      }
  else if (dir == 1) { hh = 47 - i; ww = 47 - j; }
  else if (dir == 2) { hh = j;      ww = i;      }
  else               { hh = 47 - j; ww = 47 - i; }
  float v = x[(((size_t)b * DM + c) * HH + hh) * WWD + ww];
  xs_bf[(size_t)t * DM + c] = f2bf(v);
}

// ---------------- generic f32 -> bf16 ----------------
__global__ __launch_bounds__(256) void cvt_bf16(const float* __restrict__ s,
                                                unsigned short* __restrict__ d, int n) {
  int i = blockIdx.x * 256 + threadIdx.x;
  if (i < n) d[i] = f2bf(s[i]);
}

// ---------------- bf16 WMMA GEMM:  C[M,N] = A[M,K] * W[N,K]^T ----------------
// One wave computes a (16*MT) x (16*NT) tile. K is a template constant; the
// K-loop is fully unrolled so the 2-deep ping-pong fragment buffers have
// compile-time indices and stay in VGPRs (no scratch/LDS spills).
template <int MT, int NT, int K>
__global__ __launch_bounds__(128) void wmma_gemm_bf16(
    const unsigned short* __restrict__ A,
    const unsigned short* __restrict__ Wt,
    float* __restrict__ Cm,
    int M, int N) {
  const int lane = threadIdx.x & 31;
  const int wave = threadIdx.x >> 5;
  const int m0   = blockIdx.x * (16 * MT);
  const int n0   = (blockIdx.y * (blockDim.x >> 5) + wave) * (16 * NT);
  if (m0 >= M || n0 >= N) return;                 // wave-uniform: EXEC stays all-1s

  const int r15  = lane & 15;                      // row (A) / col (B,C) within tile
  const int koff = (lane >> 4) << 3;               // K half: 0 or 8 (ISA 7.12.2 layout)

  const unsigned short* arow[MT];
  const unsigned short* wrow[NT];
#pragma unroll
  for (int i = 0; i < MT; ++i) arow[i] = A  + (size_t)(m0 + i * 16 + r15) * K + koff;
#pragma unroll
  for (int t = 0; t < NT; ++t) wrow[t] = Wt + (size_t)(n0 + t * 16 + r15) * K + koff;

  v8f acc[MT][NT];
#pragma unroll
  for (int i = 0; i < MT; ++i)
#pragma unroll
    for (int t = 0; t < NT; ++t) acc[i][t] = 0.0f;

  constexpr int KSTEPS = K / 32;
  BFfrag a[2][MT], b[2][NT];

  // prologue: step 0 into buffer 0
#pragma unroll
  for (int i = 0; i < MT; ++i) {
    a[0][i].h[0] = *(const v8u16*)(arow[i]);
    a[0][i].h[1] = *(const v8u16*)(arow[i] + 16);
  }
#pragma unroll
  for (int t = 0; t < NT; ++t) {
    b[0][t].h[0] = *(const v8u16*)(wrow[t]);
    b[0][t].h[1] = *(const v8u16*)(wrow[t] + 16);
  }

#pragma unroll
  for (int s = 0; s < KSTEPS; ++s) {
    const int cur = s & 1;                         // compile-time after unroll
    const int nxt = cur ^ 1;
    if (s + 1 < KSTEPS) {                          // prefetch next K-step fragments
      const int kn = (s + 1) * 32;
#pragma unroll
      for (int i = 0; i < MT; ++i) {
        a[nxt][i].h[0] = *(const v8u16*)(arow[i] + kn);
        a[nxt][i].h[1] = *(const v8u16*)(arow[i] + kn + 16);
      }
#pragma unroll
      for (int t = 0; t < NT; ++t) {
        b[nxt][t].h[0] = *(const v8u16*)(wrow[t] + kn);
        b[nxt][t].h[1] = *(const v8u16*)(wrow[t] + kn + 16);
      }
    }
#pragma unroll
    for (int i = 0; i < MT; ++i)
#pragma unroll
      for (int t = 0; t < NT; ++t)
        acc[i][t] = __builtin_amdgcn_wmma_f32_16x16x32_bf16(
            false, a[cur][i].bf, false, b[cur][t].bf, (short)0, acc[i][t], false, false);
  }

  // C/D layout: lane -> N = lane&15 ; VGPR r -> M = r + 8*(lane>>4)
#pragma unroll
  for (int i = 0; i < MT; ++i) {
#pragma unroll
    for (int t = 0; t < NT; ++t) {
      float* cp = Cm + (size_t)(m0 + i * 16 + ((lane >> 4) << 3)) * N + n0 + t * 16 + r15;
#pragma unroll
      for (int r = 0; r < 8; ++r) cp[(size_t)r * N] = acc[i][t][r];
    }
  }
}

// ---------------- 2) depthwise causal conv(4) + bias + SiLU ----------------
__global__ __launch_bounds__(256) void conv_silu(const float* __restrict__ xz,
                                                 const float* __restrict__ conv_w,
                                                 const float* __restrict__ conv_b,
                                                 float* __restrict__ xc,
                                                 unsigned short* __restrict__ xcb) {
  int idx = blockIdx.x * 256 + threadIdx.x;            // NSEQ*LSEQ*DI
  int d   = idx & (DI - 1);
  int t   = idx >> 9;
  int l   = t % LSEQ;
  int seq = t / LSEQ;
  size_t base = (size_t)seq * LSEQ;
  float acc = conv_b[d];
#pragma unroll
  for (int j = 0; j < 4; ++j) {
    int ls = l - 3 + j;
    if (ls >= 0) acc += conv_w[d * 4 + j] * xz[(base + ls) * 1024 + d];
  }
  float v = siluf(acc);
  xc[(size_t)t * DI + d]  = v;
  xcb[(size_t)t * DI + d] = f2bf(v);
}

// ---------------- 4) dt = softplus(x_dbl[:,:16] @ W_dt^T + b_dt) ----------------
__global__ __launch_bounds__(256) void dt_proj(const float* __restrict__ xdbl,
                                               const float* __restrict__ W_dt,
                                               const float* __restrict__ b_dt,
                                               float* __restrict__ dtb) {
  int idx = blockIdx.x * 256 + threadIdx.x;            // NSEQ*LSEQ*DI
  int d   = idx & (DI - 1);
  int t   = idx >> 9;
  const float* xr = xdbl + (size_t)t * XDW;
  float acc = b_dt[d];
#pragma unroll
  for (int r = 0; r < DTR; ++r) acc += xr[r] * W_dt[d * DTR + r];
  dtb[(size_t)t * DI + d] = (acc > 20.0f) ? acc : log1pf(__expf(acc));
}

// ---------------- 5a) chunked scan, pass 1: partial states + dt sums ----------------
__global__ __launch_bounds__(256) void scan_pass1(const float* __restrict__ dtb,
                                                  const float* __restrict__ xc,
                                                  const float* __restrict__ xdbl,
                                                  const float* __restrict__ A_log,
                                                  float* __restrict__ hpart,
                                                  float* __restrict__ dtsum) {
  int tid  = blockIdx.x * 256 + threadIdx.x;           // NSEQ*NCH*DI
  int d    = tid & (DI - 1);
  int ch   = (tid >> 9) & (NCH - 1);
  int seq  = tid >> 13;
  float Aa[NSTT];
#pragma unroll
  for (int n = 0; n < NSTT; ++n) Aa[n] = -__expf(A_log[d * NSTT + n]);
  float h[NSTT];
#pragma unroll
  for (int n = 0; n < NSTT; ++n) h[n] = 0.0f;
  float dts = 0.0f;
  int l0 = ch * CHUNK;
  for (int i = 0; i < CHUNK; ++i) {
    size_t row = (size_t)seq * LSEQ + l0 + i;
    if (i + PFD < CHUNK) {                              // SW prefetch: strided rows
      size_t prow = row + PFD;
      __builtin_prefetch(dtb  + prow * DI + d, 0, 1);
      __builtin_prefetch(xc   + prow * DI + d, 0, 1);
      __builtin_prefetch(xdbl + prow * XDW + DTR, 0, 1);
    }
    float dtv = dtb[row * DI + d];
    float u   = xc [row * DI + d];
    float Bv[NSTT];
    const float4* bp = (const float4*)(xdbl + row * XDW + DTR);
#pragma unroll
    for (int q = 0; q < 4; ++q) ((float4*)Bv)[q] = bp[q];
    dts += dtv;
    float du = dtv * u;
#pragma unroll
    for (int n = 0; n < NSTT; ++n)
      h[n] = h[n] * __expf(dtv * Aa[n]) + du * Bv[n];
  }
  float* hp = hpart + (size_t)tid * NSTT;
#pragma unroll
  for (int n = 0; n < NSTT; ++n) hp[n] = h[n];
  dtsum[tid] = dts;
}

// ---------------- 5b) serial carry across the 16 chunks ----------------
__global__ __launch_bounds__(256) void scan_carry(const float* __restrict__ hpart,
                                                  const float* __restrict__ dtsum,
                                                  const float* __restrict__ A_log,
                                                  float* __restrict__ hin) {
  int tid = blockIdx.x * 256 + threadIdx.x;            // NSEQ*DI
  int d   = tid & (DI - 1);
  int seq = tid >> 9;
  float Aa[NSTT];
#pragma unroll
  for (int n = 0; n < NSTT; ++n) Aa[n] = -__expf(A_log[d * NSTT + n]);
  float hr[NSTT];
#pragma unroll
  for (int n = 0; n < NSTT; ++n) hr[n] = 0.0f;
  for (int c = 0; c < NCH; ++c) {
    size_t bi = ((size_t)(seq * NCH + c) * DI + d);
    float* ho = hin + bi * NSTT;
    const float* hp = hpart + bi * NSTT;
    float dts = dtsum[bi];
#pragma unroll
    for (int n = 0; n < NSTT; ++n) {
      ho[n] = hr[n];
      hr[n] = hp[n] + __expf(Aa[n] * dts) * hr[n];
    }
  }
}

// ---------------- 5c) pass 2: rescan with carried state, fuse gating, emit bf16 ----------------
__global__ __launch_bounds__(256) void scan_pass2(const float* __restrict__ hin,
                                                  const float* __restrict__ dtb,
                                                  const float* __restrict__ xc,
                                                  const float* __restrict__ xdbl,
                                                  const float* __restrict__ xz,
                                                  const float* __restrict__ A_log,
                                                  const float* __restrict__ Dp,
                                                  unsigned short* __restrict__ ybf) {
  int tid  = blockIdx.x * 256 + threadIdx.x;           // NSEQ*NCH*DI
  int d    = tid & (DI - 1);
  int ch   = (tid >> 9) & (NCH - 1);
  int seq  = tid >> 13;
  float Aa[NSTT];
#pragma unroll
  for (int n = 0; n < NSTT; ++n) Aa[n] = -__expf(A_log[d * NSTT + n]);
  float h[NSTT];
  const float* hp = hin + (size_t)tid * NSTT;
#pragma unroll
  for (int n = 0; n < NSTT; ++n) h[n] = hp[n];
  float Dd = Dp[d];
  int l0 = ch * CHUNK;
  for (int i = 0; i < CHUNK; ++i) {
    size_t row = (size_t)seq * LSEQ + l0 + i;
    if (i + PFD < CHUNK) {                              // SW prefetch: strided rows
      size_t prow = row + PFD;
      __builtin_prefetch(dtb  + prow * DI + d, 0, 1);
      __builtin_prefetch(xc   + prow * DI + d, 0, 1);
      __builtin_prefetch(xdbl + prow * XDW + DTR, 0, 1);
      __builtin_prefetch(xz   + prow * 1024 + DI + d, 0, 1);
    }
    float dtv = dtb[row * DI + d];
    float u   = xc [row * DI + d];
    float Bv[NSTT], Cv[NSTT];
    const float4* bp = (const float4*)(xdbl + row * XDW + DTR);
#pragma unroll
    for (int q = 0; q < 4; ++q) { ((float4*)Bv)[q] = bp[q]; ((float4*)Cv)[q] = bp[4 + q]; }
    float du = dtv * u;
    float y  = 0.0f;
#pragma unroll
    for (int n = 0; n < NSTT; ++n) {
      h[n] = h[n] * __expf(dtv * Aa[n]) + du * Bv[n];
      y   += h[n] * Cv[n];
    }
    float z = xz[row * 1024 + DI + d];
    ybf[row * DI + d] = f2bf((y + u * Dd) * siluf(z));
  }
}

// ---------------- 7) combine 4 directions back into (b,c,h,w) ----------------
__global__ __launch_bounds__(256) void combine_out(const float* __restrict__ ob,
                                                   float* __restrict__ out) {
  int idx = blockIdx.x * 256 + threadIdx.x;            // NB*DM*HH*WWD
  int w = idx % WWD;
  int h = (idx / WWD) % HH;
  int c = (idx / (WWD * HH)) % DM;
  int b = idx / (WWD * HH * DM);
  int l0 = h * WWD + w;
  int l1 = (47 - h) * WWD + (47 - w);
  int l2 = w * WWD + h;
  int l3 = (47 - w) * WWD + (47 - h);
  float v = ob[((size_t)(0  + b) * LSEQ + l0) * DM + c]
          + ob[((size_t)(4  + b) * LSEQ + l1) * DM + c]
          + ob[((size_t)(8  + b) * LSEQ + l2) * DM + c]
          + ob[((size_t)(12 + b) * LSEQ + l3) * DM + c];
  out[idx] = v;
}

// ---------------- host: workspace layout + launch graph ----------------
extern "C" void kernel_launch(void* const* d_in, const int* in_sizes, int n_in,
                              void* d_out, int out_size, void* d_ws, size_t ws_size,
                              hipStream_t stream) {
  const float* x      = (const float*)d_in[0];
  const float* W_in   = (const float*)d_in[1];
  const float* conv_w = (const float*)d_in[2];
  const float* conv_b = (const float*)d_in[3];
  const float* W_x    = (const float*)d_in[4];
  const float* W_dt   = (const float*)d_in[5];
  const float* b_dt   = (const float*)d_in[6];
  const float* A_log  = (const float*)d_in[7];
  const float* Dp     = (const float*)d_in[8];
  const float* W_out  = (const float*)d_in[9];
  float* out = (float*)d_out;

  char* ws = (char*)d_ws;
  size_t off = 0;
  auto alloc = [&](size_t bytes) -> char* {
    char* p = ws + off;
    off = (off + bytes + 255) & ~(size_t)255;
    return p;
  };

  unsigned short* xs_bf  = (unsigned short*)alloc((size_t)MROW * DM * 2);
  unsigned short* win_bf = (unsigned short*)alloc((size_t)1024 * DM * 2);
  unsigned short* wx_bf  = (unsigned short*)alloc((size_t)XDW * DI * 2);
  unsigned short* wo_bf  = (unsigned short*)alloc((size_t)DM * DI * 2);
  float* xz    = (float*)alloc((size_t)MROW * 1024 * 4);
  float* xc    = (float*)alloc((size_t)MROW * DI * 4);
  unsigned short* xcb = (unsigned short*)alloc((size_t)MROW * DI * 2);
  float* xdbl  = (float*)alloc((size_t)MROW * XDW * 4);
  float* dtb   = (float*)alloc((size_t)MROW * DI * 4);
  float* hpart = (float*)alloc((size_t)NSEQ * NCH * DI * NSTT * 4);
  float* dtsum = (float*)alloc((size_t)NSEQ * NCH * DI * 4);
  float* hin   = (float*)alloc((size_t)NSEQ * NCH * DI * NSTT * 4);
  unsigned short* ybf = (unsigned short*)alloc((size_t)MROW * DI * 2);
  float* outb  = (float*)alloc((size_t)MROW * DM * 4);
  (void)ws_size; (void)n_in; (void)in_sizes; (void)out_size;

  // 0) gather + bf16 conversions
  gather_xs<<<(MROW * DM) / 256, 256, 0, stream>>>(x, xs_bf);
  cvt_bf16<<<(1024 * DM + 255) / 256, 256, 0, stream>>>(W_in,  win_bf, 1024 * DM);
  cvt_bf16<<<(XDW * DI + 255) / 256, 256, 0, stream>>>(W_x,   wx_bf,  XDW * DI);
  cvt_bf16<<<(DM * DI + 255) / 256, 256, 0, stream>>>(W_out, wo_bf,  DM * DI);

  // 1) in-projection: xz[M,1024] = xs[M,256] @ W_in^T   (WMMA bf16, 32x64/wave)
  wmma_gemm_bf16<2, 4, DM><<<dim3(MROW / 32, 1024 / 256), 128, 0, stream>>>(
      xs_bf, win_bf, xz, MROW, 1024);

  // 2) depthwise conv + SiLU
  conv_silu<<<(MROW * DI) / 256, 256, 0, stream>>>(xz, conv_w, conv_b, xc, xcb);

  // 3) x_dbl[M,48] = xc[M,512] @ W_x^T   (WMMA bf16, 32x48/wave)
  wmma_gemm_bf16<2, 3, DI><<<dim3(MROW / 32, 1), 32, 0, stream>>>(
      xcb, wx_bf, xdbl, MROW, XDW);

  // 4) dt projection + softplus
  dt_proj<<<(MROW * DI) / 256, 256, 0, stream>>>(xdbl, W_dt, b_dt, dtb);

  // 5) chunked selective scan (two-pass, 131k-way parallel)
  scan_pass1<<<(NSEQ * NCH * DI) / 256, 256, 0, stream>>>(dtb, xc, xdbl, A_log, hpart, dtsum);
  scan_carry<<<(NSEQ * DI) / 256, 256, 0, stream>>>(hpart, dtsum, A_log, hin);
  scan_pass2<<<(NSEQ * NCH * DI) / 256, 256, 0, stream>>>(hin, dtb, xc, xdbl, xz, A_log, Dp, ybf);

  // 6) out-projection: outb[M,256] = ybf[M,512] @ W_out^T   (WMMA bf16, 32x64/wave)
  wmma_gemm_bf16<2, 4, DI><<<dim3(MROW / 32, 1), 128, 0, stream>>>(
      ybf, wo_bf, outb, MROW, DM);

  // 7) merge the 4 directional outputs
  combine_out<<<(NB * DM * HH * WWD) / 256, 256, 0, stream>>>(outb, out);
}
